// MultiHeadSelfAttention_13443247636807
// MI455X (gfx1250) — compile-verified, matrix operand order checked by reading, not run on hardware
//
#include <hip/hip_runtime.h>
#include <hip/hip_bf16.h>
#include <math.h>

#define NB 2
#define NS 2048
#define NE 1024
#define NH 16
#define ND 64

typedef __attribute__((ext_vector_type(16))) __bf16 v16bf;
typedef __attribute__((ext_vector_type(8)))  float  v8f;

union FragBF {
  v16bf v;
  uint4 q[2];
};

__device__ __forceinline__ unsigned short f2bf(float f) {
  unsigned u = __float_as_uint(f);
  u += 0x7FFFu + ((u >> 16) & 1u);   // round-to-nearest-even
  return (unsigned short)(u >> 16);
}

// Async copy of 16 bytes global -> LDS (tracked by ASYNCcnt). The instruction
// offset is added to BOTH the LDS and global addresses (ISA 15.18.3).
__device__ __forceinline__ void async_b128(unsigned lds_off, const void* gptr) {
  unsigned long long ga = (unsigned long long)(size_t)gptr;
  asm volatile("global_load_async_to_lds_b128 %0, %1, off"
               :: "v"(lds_off), "v"(ga) : "memory");
}
__device__ __forceinline__ void async_b128_off16(unsigned lds_off, const void* gptr) {
  unsigned long long ga = (unsigned long long)(size_t)gptr;
  asm volatile("global_load_async_to_lds_b128 %0, %1, off offset:16"
               :: "v"(lds_off), "v"(ga) : "memory");
}
__device__ __forceinline__ void wait_async0() {
  asm volatile("s_wait_asynccnt 0x0" ::: "memory");
}

// ---------------------------------------------------------------------------
// f32 -> bf16 bulk conversion (vectorized x4)
// ---------------------------------------------------------------------------
__global__ void cvt_f32_bf16_kernel(const float* __restrict__ src,
                                    unsigned short* __restrict__ dst, int n4) {
  int i = blockIdx.x * blockDim.x + threadIdx.x;
  if (i < n4) {
    float4 f = ((const float4*)src)[i];
    union { unsigned short s[4]; uint2 u; } t;
    t.s[0] = f2bf(f.x); t.s[1] = f2bf(f.y);
    t.s[2] = f2bf(f.z); t.s[3] = f2bf(f.w);
    ((uint2*)dst)[i] = t.u;
  }
}

// ---------------------------------------------------------------------------
// Tiled bf16 GEMM, 128x128 workgroup tile, 8 waves (2x4), K-step 32.
// A tile staged via async global->LDS DMA; B tile transposed through VGPRs.
// MODE 0: QKV projection epilogue (bias + RoPE + scatter q/k/v bf16)
// MODE 1: out-projection epilogue (bias + f32 store)
// ---------------------------------------------------------------------------
template <int MODE>
__global__ __launch_bounds__(256) void gemm_bf16_kernel(
    const unsigned short* __restrict__ A,   // [M][K] bf16 row-major
    const unsigned short* __restrict__ Bm,  // [K][N] bf16 row-major
    const float* __restrict__ bias,         // [N]
    int M, int N, int K,
    unsigned short* __restrict__ q_ws,      // MODE 0: [b][h][s][d] bf16
    unsigned short* __restrict__ k_ws,
    unsigned short* __restrict__ v_ws,
    float* __restrict__ outf)               // MODE 1: [M][N] f32
{
  __shared__ unsigned short As[128][40];    // A tile [m][k], pad to 40
  __shared__ unsigned short Bs[128][40];    // B tile transposed [n][k]

  const int tid  = threadIdx.x;
  const int lane = tid & 31;
  const int wid  = tid >> 5;
  const int wm   = (wid >> 2) * 64;
  const int wn   = (wid & 3) * 32;
  const int l15  = lane & 15;
  const int hv   = lane >> 4;
  const int cb   = hv * 8;
  const int tM   = blockIdx.x * 128;
  const int tN   = blockIdx.y * 128;

  const v8f zero = {0.f, 0.f, 0.f, 0.f, 0.f, 0.f, 0.f, 0.f};
  v8f acc[4][2];
#pragma unroll
  for (int i = 0; i < 4; ++i)
#pragma unroll
    for (int j = 0; j < 2; ++j) acc[i][j] = zero;

  const int arow = tid >> 1, acol = (tid & 1) * 16;
  const int brow = tid >> 3, bcol = (tid & 7) * 16;

  for (int k0 = 0; k0 < K; k0 += 32) {
    __syncthreads();   // protect previous iteration's LDS reads
    {  // A tile: async DMA, 32 bytes per thread (2 x b128)
      const unsigned lds0 = (unsigned)(size_t)&As[arow][acol];
      const void* ga = (const void*)(A + (size_t)(tM + arow) * K + k0 + acol);
      async_b128(lds0, ga);
      async_b128_off16(lds0, ga);
    }
    {  // B tile: load + transpose into LDS: Bs[n][k]
      const uint4* src = (const uint4*)(Bm + (size_t)(k0 + brow) * N + tN + bcol);
      union { uint4 u[2]; unsigned short s[16]; } t;
      t.u[0] = src[0];
      t.u[1] = src[1];
#pragma unroll
      for (int i = 0; i < 16; ++i) Bs[bcol + i][brow] = t.s[i];
    }
    if (k0 + 32 < K) {  // global_prefetch_b8 for next K-step of B
      __builtin_prefetch((const void*)(Bm + (size_t)(k0 + 32 + brow) * N + tN + bcol), 0, 3);
    }
    wait_async0();
    __syncthreads();

    FragBF af[4], bf[2];
#pragma unroll
    for (int mt = 0; mt < 4; ++mt) {
      const unsigned short* p = &As[wm + mt * 16 + l15][0];
      af[mt].q[0] = *(const uint4*)(p + cb);
      af[mt].q[1] = *(const uint4*)(p + 16 + cb);
    }
#pragma unroll
    for (int nt = 0; nt < 2; ++nt) {
      const unsigned short* p = &Bs[wn + nt * 16 + l15][0];
      bf[nt].q[0] = *(const uint4*)(p + cb);
      bf[nt].q[1] = *(const uint4*)(p + 16 + cb);
    }
#pragma unroll
    for (int mt = 0; mt < 4; ++mt)
#pragma unroll
      for (int nt = 0; nt < 2; ++nt)
        acc[mt][nt] = __builtin_amdgcn_wmma_f32_16x16x32_bf16(
            false, af[mt].v, false, bf[nt].v, (short)0, acc[mt][nt], false, false);
  }

  // ---------------- epilogue ----------------
#pragma unroll
  for (int mt = 0; mt < 4; ++mt) {
#pragma unroll
    for (int nt = 0; nt < 2; ++nt) {
      const int n  = tN + wn + nt * 16 + l15;
      const float bn = bias[n];
#pragma unroll
      for (int r = 0; r < 8; ++r) {
        const int m = tM + wm + mt * 16 + r + 8 * hv;
        float vv = acc[mt][nt][r] + bn;
        if (MODE == 0) {
          const int sec = n >> 10;        // 0:q 1:k 2:v
          const int col = n & (NE - 1);
          const int hh  = col >> 6;
          const int dd  = col & (ND - 1);
          const int ss  = m & (NS - 1);
          const int bb  = m >> 11;
          if (sec < 2) {                  // RoPE on q and k
            float partner = __shfl_xor(vv, 1, 32);       // value at d^1
            float rot = (dd & 1) ? partner : -partner;   // rotate_half
            float inv = __expf(-(float)(dd >> 1) * (9.210340371976184f / 32.f));
            float ang = (float)ss * inv;
            float sn, cs;
            __sincosf(ang, &sn, &cs);
            vv = vv * cs + rot * sn;
            if (sec == 0) vv *= 0.125f;   // fold 1/sqrt(64) into q
          }
          unsigned short* dst = (sec == 0) ? q_ws : (sec == 1) ? k_ws : v_ws;
          dst[(((size_t)(bb * NH + hh)) * NS + ss) * ND + dd] = f2bf(vv);
        } else {
          outf[(size_t)m * N + n] = vv;
        }
      }
    }
  }
}

// ---------------------------------------------------------------------------
// Flash attention: one block per (b, h, 128 q-rows); 8 waves, 16 q-rows each.
// K chunk async-DMA'd to LDS once per block (was 8x redundant global reads);
// V chunk transposed through VGPRs; online softmax; all GEMMs via WMMA bf16.
// ---------------------------------------------------------------------------
__global__ __launch_bounds__(256) void attn_kernel(
    const unsigned short* __restrict__ q_ws,
    const unsigned short* __restrict__ k_ws,
    const unsigned short* __restrict__ v_ws,
    unsigned short* __restrict__ attn_bf)   // [b*s][h*d] bf16
{
  __shared__ unsigned short Kc[32][72];       // K chunk [kpos][d], padded
  __shared__ unsigned short Vt[64][40];       // V chunk transposed: [d][kpos]
  __shared__ unsigned short Pb[8][16][40];    // per-wave P tile [m][j]

  const int tid  = threadIdx.x;
  const int lane = tid & 31;
  const int wid  = tid >> 5;
  const int l15  = lane & 15;
  const int hv   = lane >> 4;
  const int cb   = hv * 8;
  const int qb   = blockIdx.x * 128;
  const int h    = blockIdx.y;
  const int b    = blockIdx.z;
  const size_t bh = (size_t)(b * NH + h) * NS * ND;
  const unsigned short* qp = q_ws + bh;
  const unsigned short* kp = k_ws + bh;
  const unsigned short* vp = v_ws + bh;
  const int qr = qb + wid * 16;

  // Q fragments for this wave's 16 rows (d in [0,32) and [32,64))
  FragBF qf[2];
#pragma unroll
  for (int ch = 0; ch < 2; ++ch) {
    const unsigned short* p = qp + (size_t)(qr + l15) * ND + ch * 32;
    qf[ch].q[0] = *(const uint4*)(p + cb);
    qf[ch].q[1] = *(const uint4*)(p + 16 + cb);
  }

  const v8f zero = {0.f, 0.f, 0.f, 0.f, 0.f, 0.f, 0.f, 0.f};
  v8f o[4] = {zero, zero, zero, zero};
  float mrow[8], lrow[8];
#pragma unroll
  for (int r = 0; r < 8; ++r) { mrow[r] = -__builtin_inff(); lrow[r] = 0.f; }

  const int kr = tid >> 3, dc = (tid & 7) * 8;   // staging coords
  const int nchunk = qb / 32 + 4;
  for (int c = 0; c < nchunk; ++c) {
    const int kb = c * 32;
    __syncthreads();   // protect previous iteration's Kc/Vt reads
    {  // K chunk: async DMA into padded LDS rows (one b128 per thread)
      const unsigned lds0 = (unsigned)(size_t)&Kc[kr][dc];
      async_b128(lds0, (const void*)(kp + (size_t)(kb + kr) * ND + dc));
    }
    {  // V chunk: load + transpose into Vt
      union { uint4 u; unsigned short s[8]; } t;
      t.u = *(const uint4*)(vp + (size_t)(kb + kr) * ND + dc);
#pragma unroll
      for (int i = 0; i < 8; ++i) Vt[dc + i][kr] = t.s[i];
    }
    wait_async0();
    __syncthreads();   // staging complete

    // S = Q * K^T  (scale pre-folded into Q); K frags from LDS
    v8f sf[2];
#pragma unroll
    for (int jt = 0; jt < 2; ++jt) {
      sf[jt] = zero;
#pragma unroll
      for (int ch = 0; ch < 2; ++ch) {
        FragBF kf;
        const unsigned short* p = &Kc[jt * 16 + l15][ch * 32];
        kf.q[0] = *(const uint4*)(p + cb);
        kf.q[1] = *(const uint4*)(p + 16 + cb);
        sf[jt] = __builtin_amdgcn_wmma_f32_16x16x32_bf16(
            false, qf[ch].v, false, kf.v, (short)0, sf[jt], false, false);
      }
#pragma unroll
      for (int r = 0; r < 8; ++r) {   // causal mask
        const int qg = qr + r + 8 * hv;
        const int kg = kb + jt * 16 + l15;
        if (kg > qg) sf[jt][r] = -__builtin_inff();
      }
    }

    // online softmax update (row reductions within 16-lane halves)
    float alpha[8], ms[8];
#pragma unroll
    for (int r = 0; r < 8; ++r) {
      float rm = fmaxf(sf[0][r], sf[1][r]);
#pragma unroll
      for (int w = 1; w <= 8; w <<= 1) rm = fmaxf(rm, __shfl_xor(rm, w, 32));
      const float mn = fmaxf(mrow[r], rm);
      ms[r]    = (mn == -__builtin_inff()) ? 0.f : mn;
      alpha[r] = __expf(mrow[r] - ms[r]);
      mrow[r]  = mn;
    }
#pragma unroll
    for (int r = 0; r < 8; ++r) {
      float p0 = __expf(sf[0][r] - ms[r]);
      float p1 = __expf(sf[1][r] - ms[r]);
      sf[0][r] = p0;
      sf[1][r] = p1;
      float rs = p0 + p1;
#pragma unroll
      for (int w = 1; w <= 8; w <<= 1) rs += __shfl_xor(rs, w, 32);
      lrow[r] = lrow[r] * alpha[r] + rs;
#pragma unroll
      for (int nt = 0; nt < 4; ++nt) o[nt][r] *= alpha[r];
    }

    // C-frag -> A-frag re-layout of P through per-wave LDS
#pragma unroll
    for (int jt = 0; jt < 2; ++jt)
#pragma unroll
      for (int r = 0; r < 8; ++r)
        Pb[wid][r + 8 * hv][jt * 16 + l15] = f2bf(sf[jt][r]);
    __syncthreads();   // P visible for cross-lane fragment reads

    FragBF pf;
    {
      const unsigned short* p = &Pb[wid][l15][0];
      pf.q[0] = *(const uint4*)(p + cb);
      pf.q[1] = *(const uint4*)(p + 16 + cb);
    }
    // O += P * V
#pragma unroll
    for (int nt = 0; nt < 4; ++nt) {
      FragBF vf;
      const unsigned short* p = &Vt[nt * 16 + l15][0];
      vf.q[0] = *(const uint4*)(p + cb);
      vf.q[1] = *(const uint4*)(p + 16 + cb);
      o[nt] = __builtin_amdgcn_wmma_f32_16x16x32_bf16(
          false, pf.v, false, vf.v, (short)0, o[nt], false, false);
    }
  }

  // finalize and store attn output [b*s][h*64+d] bf16
#pragma unroll
  for (int nt = 0; nt < 4; ++nt)
#pragma unroll
    for (int r = 0; r < 8; ++r) {
      const int srow = qr + r + 8 * hv;
      const int col  = h * ND + nt * 16 + l15;
      attn_bf[(size_t)(b * NS + srow) * NE + col] = f2bf(o[nt][r] / lrow[r]);
    }
}

// ---------------------------------------------------------------------------
extern "C" void kernel_launch(void* const* d_in, const int* in_sizes, int n_in,
                              void* d_out, int out_size, void* d_ws, size_t ws_size,
                              hipStream_t stream) {
  const float* x     = (const float*)d_in[0];   // [2,2048,1024]
  const float* w_qkv = (const float*)d_in[1];   // [1024,3072]
  const float* b_qkv = (const float*)d_in[2];   // [3072]
  const float* w_o   = (const float*)d_in[3];   // [1024,1024]
  const float* b_o   = (const float*)d_in[4];   // [1024]
  float* out = (float*)d_out;                   // [2,2048,1024] f32

  const int BS = NB * NS;                       // 4096 rows
  unsigned short* ws = (unsigned short*)d_ws;
  unsigned short* x_bf    = ws;                               // 4096*1024
  unsigned short* wqkv_bf = x_bf    + (size_t)BS * NE;        // 1024*3072
  unsigned short* wo_bf   = wqkv_bf + (size_t)NE * 3 * NE;    // 1024*1024
  unsigned short* q_ws    = wo_bf   + (size_t)NE * NE;        // 2*16*2048*64
  unsigned short* k_ws    = q_ws    + (size_t)NB * NH * NS * ND;
  unsigned short* v_ws    = k_ws    + (size_t)NB * NH * NS * ND;
  unsigned short* attn_bf = v_ws    + (size_t)NB * NH * NS * ND;  // 4096*1024

  // 1) convert inputs to bf16
  {
    int n4 = (BS * NE) / 4;
    cvt_f32_bf16_kernel<<<(n4 + 255) / 256, 256, 0, stream>>>(x, x_bf, n4);
    n4 = (NE * 3 * NE) / 4;
    cvt_f32_bf16_kernel<<<(n4 + 255) / 256, 256, 0, stream>>>(w_qkv, wqkv_bf, n4);
    n4 = (NE * NE) / 4;
    cvt_f32_bf16_kernel<<<(n4 + 255) / 256, 256, 0, stream>>>(w_o, wo_bf, n4);
  }

  // 2) QKV projection + bias + RoPE + scatter
  {
    dim3 grid(BS / 128, (3 * NE) / 128);
    gemm_bf16_kernel<0><<<grid, 256, 0, stream>>>(
        x_bf, wqkv_bf, b_qkv, BS, 3 * NE, NE, q_ws, k_ws, v_ws, nullptr);
  }

  // 3) flash attention
  {
    dim3 grid(NS / 128, NH, NB);
    attn_kernel<<<grid, 256, 0, stream>>>(q_ws, k_ws, v_ws, attn_bf);
  }

  // 4) output projection + bias -> f32
  {
    dim3 grid(BS / 128, NE / 128);
    gemm_bf16_kernel<1><<<grid, 256, 0, stream>>>(
        attn_bf, wo_bf, b_o, BS, NE, NE, nullptr, nullptr, nullptr, out);
  }
}
